// NeighborEmbedding_4887672783509
// MI455X (gfx1250) — matrix operand order repaired and checked by reading, not compiled, not run on hardware
//
#include <hip/hip_runtime.h>
#include <hip/hip_bf16.h>
#include <cstdint>

// ---------------------------------------------------------------------------
// PointNet++-style NeighborEmbedding for MI455X (gfx1250, wave32).
// Big GEMMs (sg1: 524288x128x128 x2, sg2: 262144x256x256 x2) run on
// v_wmma_f32_16x16x32_f16. Everything else (FPS, kNN, BN stats) is LDS-heavy.
// ---------------------------------------------------------------------------

typedef __attribute__((ext_vector_type(16))) _Float16 v16h;
typedef __attribute__((ext_vector_type(8)))  float    v8f;

#define BB   32
#define NN   2048
#define S1   512
#define S2   256
#define KNB  32      // neighbors
#define EPSV 1e-5f

// ------------------------------ stem ---------------------------------------

__global__ void stem1_kernel(const float* __restrict__ x, const float* __restrict__ w1,
                             float* __restrict__ xyz, float* __restrict__ h) {
  __shared__ float sw[64 * 3];
  if (threadIdx.x < 192) sw[threadIdx.x] = w1[threadIdx.x];
  __syncthreads();
  int gid = blockIdx.x * blockDim.x + threadIdx.x;           // over B*N
  if (gid >= BB * NN) return;
  int b = gid / NN, n = gid % NN;
  float px = x[(size_t)b * 3 * NN + 0 * NN + n];
  float py = x[(size_t)b * 3 * NN + 1 * NN + n];
  float pz = x[(size_t)b * 3 * NN + 2 * NN + n];
  xyz[(size_t)gid * 3 + 0] = px;
  xyz[(size_t)gid * 3 + 1] = py;
  xyz[(size_t)gid * 3 + 2] = pz;
  float* hr = h + (size_t)gid * 64;
#pragma unroll 8
  for (int c = 0; c < 64; ++c)
    hr[c] = px * sw[c * 3] + py * sw[c * 3 + 1] + pz * sw[c * 3 + 2];
}

__global__ void stem2_kernel(const float* __restrict__ f0, const float* __restrict__ w2,
                             float* __restrict__ h) {
  __shared__ float sw[64 * 64];
  for (int i = threadIdx.x; i < 4096; i += blockDim.x) sw[i] = w2[i];
  __syncthreads();
  int gid = blockIdx.x * blockDim.x + threadIdx.x;           // over B*N*64
  if (gid >= BB * NN * 64) return;
  int row = gid >> 6, c = gid & 63;
  const float* fr = f0 + (size_t)row * 64;
  const float* wr = sw + c * 64;
  float s = 0.f;
#pragma unroll 8
  for (int d = 0; d < 64; ++d) s += fr[d] * wr[d];
  h[gid] = s;
}

// ------------------------- BN statistics -----------------------------------

__global__ void zero_kernel(float* p, int n) {
  int i = blockIdx.x * blockDim.x + threadIdx.x;
  if (i < n) p[i] = 0.f;
}

__global__ void stats_f32_kernel(const float* __restrict__ h, long long total, int C,
                                 float* __restrict__ acc) {
  __shared__ float ssum[256], ssq[256];
  if (threadIdx.x < C) { ssum[threadIdx.x] = 0.f; ssq[threadIdx.x] = 0.f; }
  __syncthreads();
  long long stride = (long long)gridDim.x * blockDim.x;
  for (long long i = blockIdx.x * (long long)blockDim.x + threadIdx.x; i < total; i += stride) {
    float v = h[i];
    int c = (int)(i % C);
    atomicAdd(&ssum[c], v);
    atomicAdd(&ssq[c], v * v);
  }
  __syncthreads();
  if (threadIdx.x < C) {
    atomicAdd(&acc[threadIdx.x], ssum[threadIdx.x]);
    atomicAdd(&acc[C + threadIdx.x], ssq[threadIdx.x]);
  }
}

__global__ void stats_f16_kernel(const _Float16* __restrict__ h, long long total, int C,
                                 float* __restrict__ acc) {
  __shared__ float ssum[256], ssq[256];
  if (threadIdx.x < C) { ssum[threadIdx.x] = 0.f; ssq[threadIdx.x] = 0.f; }
  __syncthreads();
  long long stride = (long long)gridDim.x * blockDim.x;
  for (long long i = blockIdx.x * (long long)blockDim.x + threadIdx.x; i < total; i += stride) {
    float v = (float)h[i];
    int c = (int)(i % C);
    atomicAdd(&ssum[c], v);
    atomicAdd(&ssq[c], v * v);
  }
  __syncthreads();
  if (threadIdx.x < C) {
    atomicAdd(&acc[threadIdx.x], ssum[threadIdx.x]);
    atomicAdd(&acc[C + threadIdx.x], ssq[threadIdx.x]);
  }
}

__global__ void scale_shift_kernel(const float* __restrict__ acc, float invCnt,
                                   const float* __restrict__ g, const float* __restrict__ bta,
                                   float* __restrict__ ss, int C) {
  int c = blockIdx.x * blockDim.x + threadIdx.x;
  if (c >= C) return;
  float m = acc[c] * invCnt;
  float v = acc[C + c] * invCnt - m * m;
  float rs = rsqrtf(v + EPSV);
  float sc = g[c] * rs;
  ss[c] = sc;
  ss[C + c] = bta[c] - m * sc;
}

__global__ void apply_bnrelu_f32_kernel(float* __restrict__ h, long long total, int C,
                                        const float* __restrict__ ss) {
  long long stride = (long long)gridDim.x * blockDim.x;
  for (long long i = blockIdx.x * (long long)blockDim.x + threadIdx.x; i < total; i += stride) {
    int c = (int)(i % C);
    h[i] = fmaxf(h[i] * ss[c] + ss[C + c], 0.f);
  }
}

__global__ void apply_bnrelu_f16_kernel(_Float16* __restrict__ h, long long total, int C,
                                        const float* __restrict__ ss) {
  long long stride = (long long)gridDim.x * blockDim.x;
  for (long long i = blockIdx.x * (long long)blockDim.x + threadIdx.x; i < total; i += stride) {
    int c = (int)(i % C);
    float v = (float)h[i] * ss[c] + ss[C + c];
    h[i] = (_Float16)fmaxf(v, 0.f);
  }
}

// ------------------------------ FPS ----------------------------------------
// One block per batch; coords + running min-dist live in LDS; per-step argmax
// via block tree reduction. Sequential in S by construction (matches lax.scan).

__global__ void fps_kernel(const float* __restrict__ xyz, int N, int S,
                           int* __restrict__ outIdx) {
  __shared__ float sx[2048], sy[2048], sz[2048], sd[2048];
  __shared__ float rv[256];
  __shared__ int   ri[256];
  __shared__ int   sFar;
  int b = blockIdx.x, tid = threadIdx.x;
  const int T = 256;
  for (int i = tid; i < N; i += T) {
    sx[i] = xyz[((size_t)b * N + i) * 3 + 0];
    sy[i] = xyz[((size_t)b * N + i) * 3 + 1];
    sz[i] = xyz[((size_t)b * N + i) * 3 + 2];
    sd[i] = 1e10f;
  }
  if (tid == 0) sFar = 0;
  __syncthreads();
  for (int s = 0; s < S; ++s) {
    int far = sFar;
    if (tid == 0) outIdx[b * S + s] = far;
    float cx = sx[far], cy = sy[far], cz = sz[far];
    float best = -1.f; int bi = 0x7fffffff;
    for (int i = tid; i < N; i += T) {
      float dx = sx[i] - cx, dy = sy[i] - cy, dz = sz[i] - cz;
      float d = dx * dx + dy * dy + dz * dz;
      float nd = fminf(sd[i], d);
      sd[i] = nd;
      if (nd > best) { best = nd; bi = i; }     // i ascending -> first max kept
    }
    rv[tid] = best; ri[tid] = bi;
    __syncthreads();
    for (int off = 128; off > 0; off >>= 1) {
      if (tid < off) {
        if (rv[tid + off] > rv[tid] ||
            (rv[tid + off] == rv[tid] && ri[tid + off] < ri[tid])) {
          rv[tid] = rv[tid + off]; ri[tid] = ri[tid + off];
        }
      }
      __syncthreads();
    }
    if (tid == 0) sFar = ri[0];
    __syncthreads();
  }
}

__global__ void gather_xyz_kernel(const float* __restrict__ xyz, const int* __restrict__ idx,
                                  float* __restrict__ oxyz, int N, int S) {
  int gid = blockIdx.x * blockDim.x + threadIdx.x;            // over B*S
  if (gid >= BB * S) return;
  int b = gid / S;
  int j = idx[gid];
  oxyz[(size_t)gid * 3 + 0] = xyz[((size_t)b * N + j) * 3 + 0];
  oxyz[(size_t)gid * 3 + 1] = xyz[((size_t)b * N + j) * 3 + 1];
  oxyz[(size_t)gid * 3 + 2] = xyz[((size_t)b * N + j) * 3 + 2];
}

// ------------------------------ kNN ----------------------------------------
// One thread per center; 32-entry sorted insertion list in LDS (no scratch).

__global__ void knn_kernel(const float* __restrict__ coords, const float* __restrict__ centers,
                           int N, int S, int* __restrict__ knn) {
  __shared__ float cxs[2048], cys[2048], czs[2048];
  __shared__ float bd[128 * KNB];
  __shared__ int   bi[128 * KNB];
  int b = blockIdx.y;
  for (int i = threadIdx.x; i < N; i += blockDim.x) {
    cxs[i] = coords[((size_t)b * N + i) * 3 + 0];
    cys[i] = coords[((size_t)b * N + i) * 3 + 1];
    czs[i] = coords[((size_t)b * N + i) * 3 + 2];
  }
  __syncthreads();
  int s = blockIdx.x * blockDim.x + threadIdx.x;
  if (s >= S) return;
  float qx = centers[((size_t)b * S + s) * 3 + 0];
  float qy = centers[((size_t)b * S + s) * 3 + 1];
  float qz = centers[((size_t)b * S + s) * 3 + 2];
  float* ld = bd + threadIdx.x * KNB;
  int*   li = bi + threadIdx.x * KNB;
  for (int k = 0; k < KNB; ++k) { ld[k] = 3.4e38f; li[k] = 0; }
  for (int j = 0; j < N; ++j) {
    float dx = cxs[j] - qx, dy = cys[j] - qy, dz = czs[j] - qz;
    float d = dx * dx + dy * dy + dz * dz;
    if (d < ld[KNB - 1]) {
      int p = KNB - 1;
      while (p > 0 && ld[p - 1] > d) { ld[p] = ld[p - 1]; li[p] = li[p - 1]; --p; }
      ld[p] = d; li[p] = j;
    }
  }
  int* out = knn + ((size_t)b * S + s) * KNB;
  for (int k = 0; k < KNB; ++k) out[k] = li[k];
}

// --------------------- aggregate build (rel || center) ----------------------

__global__ void build_agg_kernel(const float* __restrict__ feat, const int* __restrict__ fpsIdx,
                                 const int* __restrict__ knnIdx, _Float16* __restrict__ agg,
                                 int N, int S, int D, long long total) {
  long long gid = blockIdx.x * (long long)blockDim.x + threadIdx.x;  // over B*S*K*D
  if (gid >= total) return;
  int d = (int)(gid % D);
  long long t = gid / D;
  int k = (int)(t % KNB); t /= KNB;
  int s = (int)(t % S);
  int b = (int)(t / S);
  int ci = fpsIdx[b * S + s];
  int nb = knnIdx[((size_t)b * S + s) * KNB + k];
  float fc = feat[((size_t)b * N + ci) * D + d];
  float fn = feat[((size_t)b * N + nb) * D + d];
  size_t row = ((size_t)(b * S + s)) * KNB + k;
  agg[row * (2 * D) + d]     = (_Float16)(fn - fc);
  agg[row * (2 * D) + D + d] = (_Float16)fc;
}

__global__ void f32_to_f16_kernel(const float* __restrict__ src, _Float16* __restrict__ dst, int n) {
  int i = blockIdx.x * blockDim.x + threadIdx.x;
  if (i < n) dst[i] = (_Float16)src[i];
}

// ------------------------- WMMA GEMM (f16 -> f32 acc) -----------------------
// Out[m][n] = sum_k A[m][k] * W[n][k].  One wave per 16x16 tile.
// A/B fragments in contiguous k order (k = 16*hi + s) for both (16-bit A 16x32, B 32x16).

__global__ void wmma_gemm_kernel(const _Float16* __restrict__ A, const _Float16* __restrict__ Bw,
                                 _Float16* __restrict__ Out, int M, int Kd, int Nc) {
  int wave = (blockIdx.x * blockDim.x + threadIdx.x) >> 5;
  int lane = threadIdx.x & 31;
  int ntiles = Nc >> 4;
  int tm = wave / ntiles;
  int tn = wave % ntiles;
  if (tm * 16 >= M) return;                     // wave-uniform: EXEC stays full
  int lh = lane & 15, hi = lane >> 4;
  int row = tm * 16 + lh;
  int col = tn * 16 + lh;
  v8f acc = {};
  for (int kb = 0; kb < Kd; kb += 32) {
    union { v16h h; unsigned int u[8]; } fa, fb;
    // A: lane(0-15) row=lh holds K = kb+0..15; lane(16-31) K=kb+16..31
    const unsigned int* pa = (const unsigned int*)(A + (size_t)row * Kd + kb + (hi << 4));
#pragma unroll
    for (int p = 0; p < 8; ++p) fa.u[p] = pa[p];
    // B[k][n] = Bw[n*Kd+k]: lane(0-15) col=lh holds K=kb+0..15; lane(16-31) K=kb+16..31
    const unsigned int* pb = (const unsigned int*)(Bw + (size_t)col * Kd + kb + (hi << 4));
#pragma unroll
    for (int p = 0; p < 8; ++p) fb.u[p] = pb[p];
    acc = __builtin_amdgcn_wmma_f32_16x16x32_f16(false, fa.h, false, fb.h,
                                                 (short)0, acc, false, false);
  }
#pragma unroll
  for (int r = 0; r < 8; ++r) {
    int m = tm * 16 + (hi << 3) + r;            // D layout: lanes 16-31 -> M=8..15
    Out[(size_t)m * Nc + tn * 16 + lh] = (_Float16)acc[r];
  }
}

// ----------------- BN + ReLU + max over K neighbors (+opt transpose) --------

__global__ void pool_kernel(const _Float16* __restrict__ h, const float* __restrict__ ss,
                            float* __restrict__ out, int S, int C, int transposeOut) {
  long long gid = blockIdx.x * (long long)blockDim.x + threadIdx.x;   // over B*S*C
  if (gid >= (long long)BB * S * C) return;
  int c = (int)(gid % C);
  long long t = gid / C;
  int s = (int)(t % S);
  int b = (int)(t / S);
  float sc = ss[c], sh = ss[C + c];
  const _Float16* p = h + ((size_t)(b * S + s)) * KNB * C + c;
  float m = 0.f;                                 // relu >= 0
#pragma unroll 8
  for (int k = 0; k < KNB; ++k) {
    float v = (float)p[(size_t)k * C] * sc + sh;
    m = fmaxf(m, fmaxf(v, 0.f));
  }
  if (transposeOut) out[((size_t)b * C + c) * S + s] = m;
  else              out[gid] = m;
}

// ---------------------------------------------------------------------------

extern "C" void kernel_launch(void* const* d_in, const int* in_sizes, int n_in,
                              void* d_out, int out_size, void* d_ws, size_t ws_size,
                              hipStream_t stream) {
  const float* x       = (const float*)d_in[0];
  const float* w1      = (const float*)d_in[1];
  const float* g1      = (const float*)d_in[2];
  const float* b1      = (const float*)d_in[3];
  const float* w2      = (const float*)d_in[4];
  const float* g2      = (const float*)d_in[5];
  const float* b2      = (const float*)d_in[6];
  const float* sg1_w1  = (const float*)d_in[7];
  const float* sg1_g1  = (const float*)d_in[8];
  const float* sg1_b1  = (const float*)d_in[9];
  const float* sg1_w2  = (const float*)d_in[10];
  const float* sg1_g2  = (const float*)d_in[11];
  const float* sg1_b2  = (const float*)d_in[12];
  const float* sg2_w1  = (const float*)d_in[13];
  const float* sg2_g1  = (const float*)d_in[14];
  const float* sg2_b1  = (const float*)d_in[15];
  const float* sg2_w2  = (const float*)d_in[16];
  const float* sg2_g2  = (const float*)d_in[17];
  const float* sg2_b2  = (const float*)d_in[18];
  float* out = (float*)d_out;

  // ---- workspace layout (~450 MB) ----
  char* ws = (char*)d_ws;
  size_t off = 0;
  auto alloc = [&](size_t bytes) { size_t r = off; off += (bytes + 255) & ~(size_t)255; return r; };
  float*    xyz  = (float*)(ws + alloc((size_t)BB * NN * 3 * 4));
  float*    fbuf = (float*)(ws + alloc((size_t)BB * NN * 64 * 4));   // stem output f
  float*    htmp = (float*)(ws + alloc((size_t)BB * NN * 64 * 4));   // stem intermediate
  float*    acc  = (float*)(ws + alloc(512 * 4));                    // per-ch sum/sumsq
  float*    ssb  = (float*)(ws + alloc(512 * 4));                    // folded scale/shift
  int*      fps1 = (int*)  (ws + alloc((size_t)BB * S1 * 4));
  int*      knn1 = (int*)  (ws + alloc((size_t)BB * S1 * KNB * 4));
  float*    nx1  = (float*)(ws + alloc((size_t)BB * S1 * 3 * 4));
  int*      fps2 = (int*)  (ws + alloc((size_t)BB * S2 * 4));
  int*      knn2 = (int*)  (ws + alloc((size_t)BB * S2 * KNB * 4));
  float*    nx2  = (float*)(ws + alloc((size_t)BB * S2 * 3 * 4));
  float*    f1   = (float*)(ws + alloc((size_t)BB * S1 * 128 * 4));
  _Float16* wfa  = (_Float16*)(ws + alloc(256 * 256 * 2));
  _Float16* wfb  = (_Float16*)(ws + alloc(256 * 256 * 2));
  const size_t BIG = (size_t)134217728;                              // 128 MB each
  _Float16* bigA = (_Float16*)(ws + alloc(BIG));
  _Float16* bigB = (_Float16*)(ws + alloc(BIG));
  _Float16* bigC = (_Float16*)(ws + alloc(BIG));
  (void)ws_size; (void)in_sizes; (void)n_in; (void)out_size;

  const int T = 256;
  // ================= stem =================
  stem1_kernel<<<(BB * NN + T - 1) / T, T, 0, stream>>>(x, w1, xyz, htmp);
  {
    long long tot = (long long)BB * NN * 64;
    zero_kernel<<<2, T, 0, stream>>>(acc, 512);
    stats_f32_kernel<<<1024, T, 0, stream>>>(htmp, tot, 64, acc);
    scale_shift_kernel<<<1, T, 0, stream>>>(acc, 1.f / (BB * NN), g1, b1, ssb, 64);
    apply_bnrelu_f32_kernel<<<2048, T, 0, stream>>>(htmp, tot, 64, ssb);
    stem2_kernel<<<(int)((tot + T - 1) / T), T, 0, stream>>>(htmp, w2, fbuf);
    zero_kernel<<<2, T, 0, stream>>>(acc, 512);
    stats_f32_kernel<<<1024, T, 0, stream>>>(fbuf, tot, 64, acc);
    scale_shift_kernel<<<1, T, 0, stream>>>(acc, 1.f / (BB * NN), g2, b2, ssb, 64);
    apply_bnrelu_f32_kernel<<<2048, T, 0, stream>>>(fbuf, tot, 64, ssb);
  }
  // ================= sg1 =================
  fps_kernel<<<BB, T, 0, stream>>>(xyz, NN, S1, fps1);
  gather_xyz_kernel<<<(BB * S1 + T - 1) / T, T, 0, stream>>>(xyz, fps1, nx1, NN, S1);
  knn_kernel<<<dim3(S1 / 128, BB), 128, 0, stream>>>(xyz, nx1, NN, S1, knn1);
  {
    long long tot = (long long)BB * S1 * KNB * 64;                   // agg elements / 2
    build_agg_kernel<<<(int)((tot + T - 1) / T), T, 0, stream>>>(fbuf, fps1, knn1, bigA,
                                                                 NN, S1, 64, tot);
  }
  f32_to_f16_kernel<<<(128 * 128 + T - 1) / T, T, 0, stream>>>(sg1_w1, wfa, 128 * 128);
  {
    const int M = BB * S1 * KNB;                                     // 524288
    const long long tot = (long long)M * 128;
    wmma_gemm_kernel<<<(M / 16) * (128 / 16) * 32 / T, T, 0, stream>>>(bigA, wfa, bigB, M, 128, 128);
    zero_kernel<<<2, T, 0, stream>>>(acc, 512);
    stats_f16_kernel<<<4096, T, 0, stream>>>(bigB, tot, 128, acc);
    scale_shift_kernel<<<1, T, 0, stream>>>(acc, 1.f / M, sg1_g1, sg1_b1, ssb, 128);
    apply_bnrelu_f16_kernel<<<4096, T, 0, stream>>>(bigB, tot, 128, ssb);
    f32_to_f16_kernel<<<(128 * 128 + T - 1) / T, T, 0, stream>>>(sg1_w2, wfb, 128 * 128);
    wmma_gemm_kernel<<<(M / 16) * (128 / 16) * 32 / T, T, 0, stream>>>(bigB, wfb, bigC, M, 128, 128);
    zero_kernel<<<2, T, 0, stream>>>(acc, 512);
    stats_f16_kernel<<<4096, T, 0, stream>>>(bigC, tot, 128, acc);
    scale_shift_kernel<<<1, T, 0, stream>>>(acc, 1.f / M, sg1_g2, sg1_b2, ssb, 128);
    pool_kernel<<<(int)(((long long)BB * S1 * 128 + T - 1) / T), T, 0, stream>>>(bigC, ssb, f1,
                                                                                 S1, 128, 0);
  }
  // ================= sg2 =================
  fps_kernel<<<BB, T, 0, stream>>>(nx1, S1, S2, fps2);
  gather_xyz_kernel<<<(BB * S2 + T - 1) / T, T, 0, stream>>>(nx1, fps2, nx2, S1, S2);
  knn_kernel<<<dim3(S2 / 128, BB), 128, 0, stream>>>(nx1, nx2, S1, S2, knn2);
  {
    long long tot = (long long)BB * S2 * KNB * 128;
    build_agg_kernel<<<(int)((tot + T - 1) / T), T, 0, stream>>>(f1, fps2, knn2, bigA,
                                                                 S1, S2, 128, tot);
  }
  f32_to_f16_kernel<<<(256 * 256 + T - 1) / T, T, 0, stream>>>(sg2_w1, wfa, 256 * 256);
  {
    const int M = BB * S2 * KNB;                                     // 262144
    const long long tot = (long long)M * 256;
    wmma_gemm_kernel<<<(M / 16) * (256 / 16) * 32 / T, T, 0, stream>>>(bigA, wfa, bigB, M, 256, 256);
    zero_kernel<<<2, T, 0, stream>>>(acc, 512);
    stats_f16_kernel<<<4096, T, 0, stream>>>(bigB, tot, 256, acc);
    scale_shift_kernel<<<1, T, 0, stream>>>(acc, 1.f / M, sg2_g1, sg2_b1, ssb, 256);
    apply_bnrelu_f16_kernel<<<4096, T, 0, stream>>>(bigB, tot, 256, ssb);
    f32_to_f16_kernel<<<(256 * 256 + T - 1) / T, T, 0, stream>>>(sg2_w2, wfb, 256 * 256);
    wmma_gemm_kernel<<<(M / 16) * (256 / 16) * 32 / T, T, 0, stream>>>(bigB, wfb, bigC, M, 256, 256);
    zero_kernel<<<2, T, 0, stream>>>(acc, 512);
    stats_f16_kernel<<<4096, T, 0, stream>>>(bigC, tot, 256, acc);
    scale_shift_kernel<<<1, T, 0, stream>>>(acc, 1.f / M, sg2_g2, sg2_b2, ssb, 256);
    // final: BN+ReLU+max over K, written transposed -> out[B,256,S2]
    pool_kernel<<<(int)(((long long)BB * S2 * 256 + T - 1) / T), T, 0, stream>>>(bigC, ssb, out,
                                                                                 S2, 256, 1);
  }
}
